// SimpleLSTM_326417515296
// MI455X (gfx1250) — matrix operand order repaired and checked by reading, not compiled, not run on hardware
//
#include <hip/hip_runtime.h>
#include <hip/hip_bf16.h>

// ---------------------------------------------------------------------------
// 2-layer LSTM (B=32, T=512, D=512, H=1024, O=512) for gfx1250 (CDNA5).
// bf16 WMMA (v_wmma_f32_16x16x32_bf16) everywhere; fp32 accumulate/state.
// ---------------------------------------------------------------------------

#define DEV __device__ __forceinline__

typedef __attribute__((ext_vector_type(16))) __bf16   v16bf;
typedef __attribute__((ext_vector_type(8)))  float    v8f;
typedef __attribute__((ext_vector_type(4)))  unsigned int u32x4;

union ABFrag { u32x4 u[2]; v16bf v; };

DEV v8f wmma_bf16(v16bf a, v16bf b, v8f c) {
  return __builtin_amdgcn_wmma_f32_16x16x32_bf16(
      /*neg_a=*/false, a, /*neg_b=*/false, b,
      /*c_mod=*/(short)0, c, /*reuse_a=*/false, /*reuse_b=*/false);
}

DEV float sigf(float x) { return 1.0f / (1.0f + __expf(-x)); }

// Model constants
enum : int { Bz = 32, Tz = 512, Dz = 512, Hz = 1024, Oz = 512, Gz = 4096 };

// ---------------------------------------------------------------------------
// fp32 -> bf16 elementwise convert (row-major)
// ---------------------------------------------------------------------------
__global__ void k_f32_to_bf16(const float* __restrict__ s,
                              __hip_bfloat16* __restrict__ d, int n) {
  int i = blockIdx.x * blockDim.x + threadIdx.x;
  if (i < n) d[i] = __float2bfloat16(s[i]);
}

__global__ void k_zero_f32(float* __restrict__ p, int n) {
  int i = blockIdx.x * blockDim.x + threadIdx.x;
  if (i < n) p[i] = 0.0f;
}

// ---------------------------------------------------------------------------
// Swizzle weight W[K,N] (fp32 row-major) into bf16 WMMA-B tiles.
// Tile (kt,nt) is 32(K)x16(N), stored as 32 lanes x 16 contiguous bf16:
//   element e of lane l maps to  K = kt*32 + (l>=16 ? 8 : 0) + (e&7) + (e>=8 ? 16 : 0),
//                                N = nt*16 + (l & 15)
// Flat dst index == tid with tid = ((kt*NT + nt)*32 + lane)*16 + e.
// ---------------------------------------------------------------------------
__global__ void k_swizzle_w(const float* __restrict__ W,
                            __hip_bfloat16* __restrict__ dst, int N, int total) {
  int tid = blockIdx.x * blockDim.x + threadIdx.x;
  if (tid >= total) return;
  int e    = tid & 15;
  int lane = (tid >> 4) & 31;
  int tile = tid >> 9;
  int NT   = N >> 4;
  int nt   = tile % NT;
  int kt   = tile / NT;
  int k = (kt << 5) + ((lane >> 4) << 3) + (e & 7) + ((e >> 3) << 4);
  int n = (nt << 4) + (lane & 15);
  dst[tid] = __float2bfloat16(W[(size_t)k * N + n]);
}

// ---------------------------------------------------------------------------
// One wave computes a 32x16 fp32 tile of A[M,K]@B[K,N] (A bf16 row-major,
// B pre-swizzled tiles). acc[0] = rows mrow..mrow+15, acc[1] = +16..+31.
// A fragment layout per ISA: lane<16 holds K{0..7,16..23}, lane>=16 holds
// K{8..15,24..31} of row (lane&15)  ->  two b128 loads per 16x32 tile.
// ---------------------------------------------------------------------------
DEV void wave_gemm32x16(const __hip_bfloat16* __restrict__ A, int lda, int mrow,
                        const __hip_bfloat16* __restrict__ Bt, int NT, int nt,
                        int K, int lane, v8f acc[2]) {
  int lrow  = lane & 15;
  int khalf = (lane >> 4) << 3;
  const __hip_bfloat16* pa0 = A + (size_t)(mrow + lrow) * lda + khalf;
  const __hip_bfloat16* pa1 = pa0 + (size_t)16 * lda;
  for (int k = 0; k < K; k += 32) {
    ABFrag a0, a1, b;
    a0.u[0] = *reinterpret_cast<const u32x4*>(pa0 + k);
    a0.u[1] = *reinterpret_cast<const u32x4*>(pa0 + k + 16);
    a1.u[0] = *reinterpret_cast<const u32x4*>(pa1 + k);
    a1.u[1] = *reinterpret_cast<const u32x4*>(pa1 + k + 16);
    const __hip_bfloat16* pb =
        Bt + ((((size_t)(k >> 5)) * NT + nt) * 32 + lane) * 16;
    b.u[0] = *reinterpret_cast<const u32x4*>(pb);
    b.u[1] = *reinterpret_cast<const u32x4*>(pb + 8);
    acc[0] = wmma_bf16(a0.v, b.v, acc[0]);
    acc[1] = wmma_bf16(a1.v, b.v, acc[1]);
  }
}

// ---------------------------------------------------------------------------
// Z0 = xb[16384,512] @ Wx0s + b0  -> bf16 Z0[16384,4096]
// grid 16384 blocks x 256 thr (8 waves); wave tile 32x16.
// ---------------------------------------------------------------------------
__global__ __launch_bounds__(256)
void k_gemm_z0(const __hip_bfloat16* __restrict__ xb,
               const __hip_bfloat16* __restrict__ Wt,
               const float* __restrict__ b0,
               __hip_bfloat16* __restrict__ Z0) {
  int lane = threadIdx.x & 31, w = threadIdx.x >> 5;
  int mt = blockIdx.x >> 5;
  int nt = ((blockIdx.x & 31) << 3) + w;
  int mrow = mt << 5;
  v8f acc[2] = {};
  wave_gemm32x16(xb, Dz, mrow, Wt, Gz / 16, nt, Dz, lane, acc);
  int col = (nt << 4) + (lane & 15);
  float bb = b0[col];
#pragma unroll
  for (int m = 0; m < 2; ++m)
#pragma unroll
    for (int r = 0; r < 8; ++r) {
      int row = mrow + (m << 4) + ((lane >> 4) << 3) + r;
      Z0[(size_t)row * Gz + col] = __float2bfloat16(acc[m][r] + bb);
    }
}

// ---------------------------------------------------------------------------
// out[b,t,:] = hall[(t*32+b),:] @ Wd + bd   (fp32 output)
// grid 2048 blocks x 256 thr; wave tile 32x16.
// ---------------------------------------------------------------------------
__global__ __launch_bounds__(256)
void k_gemm_out(const __hip_bfloat16* __restrict__ hall,
                const __hip_bfloat16* __restrict__ Wdt,
                const float* __restrict__ bd,
                float* __restrict__ out) {
  int lane = threadIdx.x & 31, w = threadIdx.x >> 5;
  int mt = blockIdx.x >> 2;
  int nt = ((blockIdx.x & 3) << 3) + w;
  int mrow = mt << 5;
  v8f acc[2] = {};
  wave_gemm32x16(hall, Hz, mrow, Wdt, Oz / 16, nt, Hz, lane, acc);
  int col = (nt << 4) + (lane & 15);
  float bb = bd[col];
#pragma unroll
  for (int m = 0; m < 2; ++m)
#pragma unroll
    for (int r = 0; r < 8; ++r) {
      int row = mrow + (m << 4) + ((lane >> 4) << 3) + r;
      int tt = row >> 5, bi = row & 31;
      out[((size_t)bi * Tz + tt) * Oz + col] = acc[m][r] + bb;
    }
}

// ---------------------------------------------------------------------------
// One LSTM cell step for one layer.
//   layer 0: z = Z0[t] + h0_prev @ Wh0          (bias folded into Z0)
//   layer 1: z = h0_t @ Wx1 + h1_prev @ Wh1 + b1
// grid: 64 blocks (one 16-col tile per gate) x 128 thr (4 waves, K-split).
// Wave partials reduced through LDS; waves 0/1 apply the gate math for the
// two 16-row halves and write c (fp32, in place) and h (bf16, ping-pong).
// ---------------------------------------------------------------------------
__global__ __launch_bounds__(128)
void k_lstm_cell(int layer,
                 const __hip_bfloat16* __restrict__ A0,   // h_prev / h0_t
                 const __hip_bfloat16* __restrict__ A1,   // h1_prev (layer1)
                 const __hip_bfloat16* __restrict__ Bt0,  // Wh0s / Wx1s
                 const __hip_bfloat16* __restrict__ Bt1,  // -- / Wh1s
                 const __hip_bfloat16* __restrict__ zadd, // Z0 + t*4096 (layer0)
                 const float* __restrict__ bias,          // b1 (layer1)
                 float* __restrict__ c,
                 __hip_bfloat16* __restrict__ hout,
                 __hip_bfloat16* __restrict__ hall)       // h1_all slice (layer1)
{
  int lane = threadIdx.x & 31, w = threadIdx.x >> 5;
  int ct = blockIdx.x;  // 0..63 : gate-local 16-column tile
  const __hip_bfloat16* A;
  const __hip_bfloat16* Bt;
  int kbase, klen;
  if (layer == 0) { A = A0; Bt = Bt0; kbase = w << 8; klen = 256; }
  else {
    klen = 512;
    if (w < 2) { A = A0; Bt = Bt0; kbase = w << 9; }
    else       { A = A1; Bt = Bt1; kbase = (w - 2) << 9; }
  }

  int lrow = lane & 15, khalf = (lane >> 4) << 3;
  const __hip_bfloat16* pa0 = A + (size_t)lrow * Hz + kbase + khalf;
  const __hip_bfloat16* pa1 = pa0 + 16 * Hz;

  v8f acc[4][2] = {};
  for (int k = 0; k < klen; k += 32) {
    ABFrag a0, a1;
    a0.u[0] = *reinterpret_cast<const u32x4*>(pa0 + k);
    a0.u[1] = *reinterpret_cast<const u32x4*>(pa0 + k + 16);
    a1.u[0] = *reinterpret_cast<const u32x4*>(pa1 + k);
    a1.u[1] = *reinterpret_cast<const u32x4*>(pa1 + k + 16);
    int kt = (kbase + k) >> 5;
#pragma unroll
    for (int g = 0; g < 4; ++g) {
      ABFrag b;
      const __hip_bfloat16* pb =
          Bt + (((size_t)kt * 256 + (g << 6) + ct) * 32 + lane) * 16;
      b.u[0] = *reinterpret_cast<const u32x4*>(pb);
      b.u[1] = *reinterpret_cast<const u32x4*>(pb + 8);
      acc[g][0] = wmma_bf16(a0.v, b.v, acc[g][0]);
      acc[g][1] = wmma_bf16(a1.v, b.v, acc[g][1]);
    }
  }

  __shared__ float red[4][4][2][8][32];  // [wave][gate][mtile][r][lane] 32KB
#pragma unroll
  for (int g = 0; g < 4; ++g)
#pragma unroll
    for (int m = 0; m < 2; ++m)
#pragma unroll
      for (int r = 0; r < 8; ++r) red[w][g][m][r][lane] = acc[g][m][r];
  __syncthreads();

  if (w < 2) {
    int m = w;                                // which 16-row half
    int col = (ct << 4) + (lane & 15);        // 0..1023 within H
    int rbase = (m << 4) + ((lane >> 4) << 3);
    for (int r = 0; r < 8; ++r) {
      int row = rbase + r;                    // batch index 0..31
      float z[4];
#pragma unroll
      for (int g = 0; g < 4; ++g) {
        z[g] = red[0][g][m][r][lane] + red[1][g][m][r][lane] +
               red[2][g][m][r][lane] + red[3][g][m][r][lane];
        if (bias) z[g] += bias[(g << 10) + col];
        if (zadd)
          z[g] += __bfloat162float(
              zadd[(size_t)row * ((size_t)Tz * Gz) + (g << 10) + col]);
      }
      float iv = sigf(z[0]), fv = sigf(z[1]);
      float gv = tanhf(z[2]), ov = sigf(z[3]);
      size_t idx = ((size_t)row << 10) + col;
      float cn = fv * c[idx] + iv * gv;
      c[idx] = cn;
      float hn = ov * tanhf(cn);
      __hip_bfloat16 hb = __float2bfloat16(hn);
      hout[idx] = hb;
      if (hall) hall[idx] = hb;
    }
  }
}

// ---------------------------------------------------------------------------
// Host launch
// ---------------------------------------------------------------------------
extern "C" void kernel_launch(void* const* d_in, const int* in_sizes, int n_in,
                              void* d_out, int out_size, void* d_ws, size_t ws_size,
                              hipStream_t stream) {
  (void)in_sizes; (void)n_in; (void)out_size; (void)ws_size;
  const float* x   = (const float*)d_in[0];  // [32,512,512]
  const float* Wx0 = (const float*)d_in[1];  // [512,4096]
  const float* Wh0 = (const float*)d_in[2];  // [1024,4096]
  const float* b0  = (const float*)d_in[3];  // [4096]
  const float* Wx1 = (const float*)d_in[4];  // [1024,4096]
  const float* Wh1 = (const float*)d_in[5];  // [1024,4096]
  const float* b1  = (const float*)d_in[6];  // [4096]
  const float* Wd  = (const float*)d_in[7];  // [1024,512]
  const float* bd  = (const float*)d_in[8];  // [512]
  float* out = (float*)d_out;                // [32,512,512]

  char* p = (char*)d_ws;
  auto carve = [&](size_t bytes) {
    char* r = p;
    p += (bytes + 255) & ~(size_t)255;
    return r;
  };
  __hip_bfloat16* xb   = (__hip_bfloat16*)carve((size_t)8388608 * 2);   // x bf16
  __hip_bfloat16* Wx0s = (__hip_bfloat16*)carve((size_t)2097152 * 2);
  __hip_bfloat16* Wh0s = (__hip_bfloat16*)carve((size_t)4194304 * 2);
  __hip_bfloat16* Wx1s = (__hip_bfloat16*)carve((size_t)4194304 * 2);
  __hip_bfloat16* Wh1s = (__hip_bfloat16*)carve((size_t)4194304 * 2);
  __hip_bfloat16* Wds  = (__hip_bfloat16*)carve((size_t)524288 * 2);
  __hip_bfloat16* Z0   = (__hip_bfloat16*)carve((size_t)67108864 * 2);  // 128MB
  __hip_bfloat16* hall = (__hip_bfloat16*)carve((size_t)16777216 * 2);  // 32MB
  float* c0            = (float*)carve((size_t)32768 * 4);
  float* c1            = (float*)carve((size_t)32768 * 4);
  __hip_bfloat16* h0b  = (__hip_bfloat16*)carve((size_t)2 * 32768 * 2); // ping-pong
  __hip_bfloat16* h1b  = (__hip_bfloat16*)carve((size_t)2 * 32768 * 2);

  // --- prep: convert x, swizzle weights, zero state (c0,c1,h0b,h1b = 512KB) ---
  k_f32_to_bf16<<<8388608 / 256, 256, 0, stream>>>(x, xb, 8388608);
  k_swizzle_w<<<2097152 / 256, 256, 0, stream>>>(Wx0, Wx0s, 4096, 2097152);
  k_swizzle_w<<<4194304 / 256, 256, 0, stream>>>(Wh0, Wh0s, 4096, 4194304);
  k_swizzle_w<<<4194304 / 256, 256, 0, stream>>>(Wx1, Wx1s, 4096, 4194304);
  k_swizzle_w<<<4194304 / 256, 256, 0, stream>>>(Wh1, Wh1s, 4096, 4194304);
  k_swizzle_w<<<524288 / 256, 256, 0, stream>>>(Wd, Wds, 512, 524288);
  k_zero_f32<<<131072 / 256, 256, 0, stream>>>(c0, 131072);

  // --- Z0 = x @ Wx0 + b0 (parallel over all B*T rows) ---
  k_gemm_z0<<<16384, 256, 0, stream>>>(xb, Wx0s, b0, Z0);

  // --- sequential recurrence: 512 steps x 2 layers ---
  for (int t = 0; t < Tz; ++t) {
    int pp = t & 1;
    k_lstm_cell<<<64, 128, 0, stream>>>(
        0, h0b + pp * 32768, nullptr, Wh0s, nullptr,
        Z0 + (size_t)t * Gz, nullptr, c0, h0b + (1 - pp) * 32768, nullptr);
    k_lstm_cell<<<64, 128, 0, stream>>>(
        1, h0b + (1 - pp) * 32768, h1b + pp * 32768, Wx1s, Wh1s,
        nullptr, b1, c1, h1b + (1 - pp) * 32768, hall + (size_t)t * 32768);
  }

  // --- output projection ---
  k_gemm_out<<<2048, 256, 0, stream>>>(hall, Wds, bd, out);
}
